// MultiScaleDeformableTransformerDecoderLayer_34737695490776
// MI455X (gfx1250) — compile-verified, matrix operand order checked by reading, not run on hardware
//
#include <hip/hip_runtime.h>
#include <hip/hip_bf16.h>
#include <math.h>

// ---------------------------------------------------------------------------
// Problem constants (from reference)
// ---------------------------------------------------------------------------
#define BB 8
#define QQ 1000
#define DD 256
#define HH 8
#define LL 4
#define PP 4
#define FF 2048
#define DH 32            // DD / HH  == wave32 lane count (exploited in sampler)
#define ROWS (BB * QQ)   // 8000, multiple of 16

static const int SH[4] = {92, 46, 23, 12};
static const int SW[4] = {92, 46, 23, 12};

typedef __attribute__((ext_vector_type(16))) _Float16 v16h;
typedef __attribute__((ext_vector_type(8)))  float    v8f;

// ---------------------------------------------------------------------------
// CDNA5 WMMA fragment loaders (wave32, 16x16x32 f16 <- f32 inputs)
//
// 16-bit A-matrix 16x32 layout (ISA 7.12.2): lane L holds row M = L%16.
//   lanes 0-15 : halves 0..7 -> K=0..7,  halves 8..15 -> K=16..23
//   lanes 16-31: halves 0..7 -> K=8..15, halves 8..15 -> K=24..31
// i.e. per lane the fragment is TWO CONTIGUOUS 8-float runs starting at
// K = kb and K = 16+kb (kb = 0 or 8) -> exactly four global_load_b128.
// B fragment uses the same pattern with N as the "row" (W[N,K] row-major,
// giving C = A @ W^T). f32 C/D: lane = column N, VGPR r = row r (+8 hi half).
// Out-of-range rows are CLAMPED by the caller and zeroed via scale==0 so no
// exec-mask divergence reaches the load pipeline.
// ---------------------------------------------------------------------------
__device__ __forceinline__ v16h load_frag_contig(const float* __restrict__ rowBase,
                                                 int k0, float scale) {
  const int lane = threadIdx.x & 31;
  const int kb = (lane & 16) ? 8 : 0;
  const float* p0 = rowBase + k0 + kb;
  const float* p1 = rowBase + k0 + 16 + kb;
  float c[16];
  *(float4*)(c + 0)  = *(const float4*)(p0);
  *(float4*)(c + 4)  = *(const float4*)(p0 + 4);
  *(float4*)(c + 8)  = *(const float4*)(p1);
  *(float4*)(c + 12) = *(const float4*)(p1 + 4);
  v16h f;
#pragma unroll
  for (int i = 0; i < 16; ++i) f[i] = (_Float16)(c[i] * scale);
  return f;
}

__device__ __forceinline__ v16h load_frag_strided(const float* __restrict__ base,
                                                  long row, long rowStride,
                                                  long colStride, int k0,
                                                  float scale) {
  const int lane = threadIdx.x & 31;
  const int kb = (lane & 16) ? 8 : 0;
  const float* p = base + row * rowStride;
  float c[16];
#pragma unroll
  for (int i = 0; i < 16; ++i) {
    int k = (i < 8) ? (kb + i) : (16 + kb + (i - 8));
    c[i] = p[(long)(k0 + k) * colStride];
  }
  v16h f;
#pragma unroll
  for (int i = 0; i < 16; ++i) f[i] = (_Float16)(c[i] * scale);
  return f;
}

__device__ __forceinline__ v8f wmma_f16(v16h a, v16h b, v8f c) {
  return __builtin_amdgcn_wmma_f32_16x16x32_f16(false, a, false, b,
                                                (short)0, c, false, false);
}

// ---------------------------------------------------------------------------
// Generic WMMA GEMM:  C[b][m][n] = act( A[b][m][k] * W[n][k] + bias[n] )
//
// Tiling for bandwidth-bound K=256/2048 GEMMs: each wave computes a 16x64
// strip (1 A-frag + 4 B-frags -> 4 WMMAs per k-step, doubling FLOP/byte vs a
// single 16x16 tile). The block's 4 waves take 4 consecutive m-tiles, so all
// waves issue identical W addresses (L0 hits). N % 64 == 0 at every call
// site; K % 32 == 0. A has arbitrary strides so mem[B,D,H,W] is consumed
// in-place as [HW x D].
// ---------------------------------------------------------------------------
__global__ void wmma_gemm(const float* __restrict__ A, long aRow, long aCol,
                          long aBatch, const float* __restrict__ W,
                          const float* __restrict__ bias,
                          float* __restrict__ C, long ldc, long cBatch,
                          int M, int N, int K, int relu) {
  const int lane = threadIdx.x & 31;
  const int wid = threadIdx.x >> 5;
  const int n0 = blockIdx.x * 64;
  const int m0 = (blockIdx.y * 4 + wid) * 16;
  const int z  = blockIdx.z;
  const float* Ab = A + (long)z * aBatch;

  const int arow = m0 + (lane & 15);
  const int rowc = arow < M ? arow : (M - 1);       // clamp, zero via scale
  const float asc = arow < M ? 1.0f : 0.0f;
  const float* Arow = Ab + (long)rowc * aRow;       // used when aCol == 1
  const float* W0 = W + (long)(n0 + (lane & 15)) * K;

  v8f acc[4] = {};
  if (aCol == 1) {
    for (int k0 = 0; k0 < K; k0 += 32) {
      __builtin_prefetch(Arow + k0 + 64, 0, 1);     // gfx1250 global_prefetch_b8
      __builtin_prefetch(W0 + k0 + 64, 0, 1);
      v16h a = load_frag_contig(Arow, k0, asc);
#pragma unroll
      for (int s = 0; s < 4; ++s) {
        v16h b = load_frag_contig(W0 + (long)s * 16 * K, k0, 1.0f);
        acc[s] = wmma_f16(a, b, acc[s]);
      }
    }
  } else {
    for (int k0 = 0; k0 < K; k0 += 32) {
      __builtin_prefetch(W0 + k0 + 64, 0, 1);
      v16h a = load_frag_strided(Ab, rowc, aRow, aCol, k0, asc);
#pragma unroll
      for (int s = 0; s < 4; ++s) {
        v16h b = load_frag_contig(W0 + (long)s * 16 * K, k0, 1.0f);
        acc[s] = wmma_f16(a, b, acc[s]);
      }
    }
  }

  const int rbase = m0 + ((lane >> 4) << 3);
#pragma unroll
  for (int s = 0; s < 4; ++s) {
    const int ncol = n0 + s * 16 + (lane & 15);
    const float bv = bias ? bias[ncol] : 0.0f;
#pragma unroll
    for (int r = 0; r < 8; ++r) {
      int row = rbase + r;
      if (row < M) {
        float v = acc[s][r] + bv;
        if (relu) v = fmaxf(v, 0.0f);
        C[(long)z * cBatch + (long)row * ldc + ncol] = v;
      }
    }
  }
}

// ---------------------------------------------------------------------------
// Flash self-attention, one wave per (b, h, 16-query tile).
// q/k live in the fused in-proj buffer (ld=512), v in its own (ld=256).
// Never materializes the 8x8x1000x1000 score tensor (saves ~0.5 GB of HBM
// traffic -> the decisive optimization at 23.3 TB/s). Per-(b,h) K/V working
// set is 256 KB * 64 = 16 MB -> stays resident in the 192 MB L2 across the
// 63 query tiles that stream it.
// ---------------------------------------------------------------------------
__global__ void flash_attn(const float* __restrict__ qk2,  // [ROWS,512] q|k
                           const float* __restrict__ vbuf, // [ROWS,256]
                           float* __restrict__ o) {        // [ROWS,256]
  __shared__ float sP[16][33];

  const int lane = threadIdx.x & 31;
  const int half = lane >> 4;          // 0: rows 0-7, 1: rows 8-15 (C layout)
  const int qt = blockIdx.x;
  const int b = blockIdx.y / HH;
  const int h = blockIdx.y % HH;
  const int q0 = qt * 16;

  const float* qb = qk2 + (long)b * QQ * 512 + h * DH;        // q columns
  const float* kb = qk2 + (long)b * QQ * 512 + 256 + h * DH;  // k columns
  const float* vb = vbuf + (long)b * QQ * 256 + h * DH;

  const int qrow = q0 + (lane & 15);
  const int qrc = qrow < QQ ? qrow : (QQ - 1);
  v16h qa = load_frag_contig(qb + (long)qrc * 512, 0,
                             qrow < QQ ? 0.17677669529f : 0.0f); // 1/sqrt(32)

  v8f O0 = {}, O1 = {};
  float mi[8], li[8];
#pragma unroll
  for (int r = 0; r < 8; ++r) { mi[r] = -1e30f; li[r] = 0.0f; }

  for (int kv0 = 0; kv0 < QQ; kv0 += 32) {
    // ---- scores: two 16x16 tiles over this 32-wide kv chunk ----
    v8f S0 = {}, S1 = {};
    const int kr0 = kv0 + (lane & 15);
    const int kr1 = kr0 + 16;
    {
      int kc0 = kr0 < QQ ? kr0 : (QQ - 1);
      int kc1 = kr1 < QQ ? kr1 : (QQ - 1);
      v16h k0f = load_frag_contig(kb + (long)kc0 * 512, 0, kr0 < QQ ? 1.0f : 0.0f);
      v16h k1f = load_frag_contig(kb + (long)kc1 * 512, 0, kr1 < QQ ? 1.0f : 0.0f);
      S0 = wmma_f16(qa, k0f, S0);
      S1 = wmma_f16(qa, k1f, S1);
    }
    // mask kv columns beyond Q, branchless (per-lane column = kv index)
#pragma unroll
    for (int r = 0; r < 8; ++r) {
      S0[r] = kr0 < QQ ? S0[r] : -1e30f;
      S1[r] = kr1 < QQ ? S1[r] : -1e30f;
    }

    // ---- online softmax (row stats via half-wave shfl reductions) ----
    float pmax[8];
#pragma unroll
    for (int r = 0; r < 8; ++r) pmax[r] = fmaxf(S0[r], S1[r]);
#pragma unroll
    for (int m = 1; m < 16; m <<= 1)
#pragma unroll
      for (int r = 0; r < 8; ++r) pmax[r] = fmaxf(pmax[r], __shfl_xor(pmax[r], m));

    float scl[8], rs[8];
#pragma unroll
    for (int r = 0; r < 8; ++r) {
      float mn = fmaxf(mi[r], pmax[r]);
      scl[r] = __expf(mi[r] - mn);
      mi[r] = mn;
      float p0 = __expf(S0[r] - mn);
      float p1 = __expf(S1[r] - mn);
      sP[r + half * 8][lane & 15]        = p0;   // transpose staging
      sP[r + half * 8][(lane & 15) + 16] = p1;
      rs[r] = p0 + p1;
    }
#pragma unroll
    for (int m = 1; m < 16; m <<= 1)
#pragma unroll
      for (int r = 0; r < 8; ++r) rs[r] += __shfl_xor(rs[r], m);
#pragma unroll
    for (int r = 0; r < 8; ++r) {
      li[r] = li[r] * scl[r] + rs[r];
      O0[r] *= scl[r];
      O1[r] *= scl[r];
    }
    __syncthreads();   // single-wave block: lowers to S_NOP-class barrier

    // ---- P (A-frag, via LDS transpose) @ V chunk ----
    v16h pa;
    {
      const int prow = lane & 15;
      const int kb2 = (lane & 16) ? 8 : 0;
#pragma unroll
      for (int i = 0; i < 16; ++i) {
        int k = (i < 8) ? (kb2 + i) : (16 + kb2 + (i - 8));
        pa[i] = (_Float16)sP[prow][k];
      }
    }
    __syncthreads();
    {
      const int kb2 = (lane & 16) ? 8 : 0;
      v16h v0f, v1f;
#pragma unroll
      for (int i = 0; i < 16; ++i) {
        int k = (i < 8) ? (kb2 + i) : (16 + kb2 + (i - 8));
        int kv = kv0 + k; if (kv > QQ - 1) kv = QQ - 1;  // masked P==0 anyway
        v0f[i] = (_Float16)vb[(long)kv * 256 + (lane & 15)];
        v1f[i] = (_Float16)vb[(long)kv * 256 + 16 + (lane & 15)];
      }
      O0 = wmma_f16(pa, v0f, O0);
      O1 = wmma_f16(pa, v1f, O1);
    }
  }

  // ---- normalize + store [B,Q,H*Dh] ----
  const int rbase = q0 + half * 8;
  const int c0 = h * DH + (lane & 15);
#pragma unroll
  for (int r = 0; r < 8; ++r) {
    int row = rbase + r;
    if (row < QQ) {
      float inv = 1.0f / li[r];
      long base = ((long)b * QQ + row) * 256;
      o[base + c0]      = O0[r] * inv;
      o[base + c0 + 16] = O1[r] * inv;
    }
  }
}

// ---------------------------------------------------------------------------
// DETR sine positional embedding; also emits qk = query + pe
// ---------------------------------------------------------------------------
__global__ void pos_embed(const float* __restrict__ query,
                          const float* __restrict__ qpos,
                          float* __restrict__ pe, float* __restrict__ qk) {
  long idx = (long)blockIdx.x * blockDim.x + threadIdx.x;
  if (idx >= (long)ROWS * DD) return;
  int ch = idx & 255;
  long bq = idx >> 8;
  int c = ch >> 7;          // coord (x or y)
  int r = ch & 127;
  int j = r & 63;
  float t = __powf(10000.0f, (float)j * (1.0f / 64.0f));
  float ang = qpos[bq * 2 + c] * 6.283185307179586f / t;
  float p = (r < 64) ? __sinf(ang) : __cosf(ang);
  pe[idx] = p;
  qk[idx] = query[idx] + p;
}

__global__ void add_vec(const float* __restrict__ a, const float* __restrict__ b,
                        float* __restrict__ c, long n) {
  long i = (long)blockIdx.x * blockDim.x + threadIdx.x;
  if (i < n) c[i] = a[i] + b[i];
}

// ---------------------------------------------------------------------------
// out = LayerNorm(a + b) * g + beta   (one 256-thread block per row)
// ---------------------------------------------------------------------------
__device__ __forceinline__ float block_sum256(float v, float* sbuf) {
#pragma unroll
  for (int m = 16; m >= 1; m >>= 1) v += __shfl_xor(v, m);
  if ((threadIdx.x & 31) == 0) sbuf[threadIdx.x >> 5] = v;
  __syncthreads();
  float t = (threadIdx.x < 8) ? sbuf[threadIdx.x] : 0.0f;
#pragma unroll
  for (int m = 4; m >= 1; m >>= 1) t += __shfl_xor(t, m);
  if (threadIdx.x == 0) sbuf[0] = t;
  __syncthreads();
  float r = sbuf[0];
  __syncthreads();
  return r;
}

__global__ void ln_res(const float* __restrict__ a, const float* __restrict__ b,
                       const float* __restrict__ g, const float* __restrict__ be,
                       float* __restrict__ out) {
  __shared__ float sbuf[8];
  long row = blockIdx.x;
  int t = threadIdx.x;
  float x = a[row * DD + t] + b[row * DD + t];
  float mean = block_sum256(x, sbuf) * (1.0f / DD);
  float d = x - mean;
  float var = block_sum256(d * d, sbuf) * (1.0f / DD);
  out[row * DD + t] = d * rsqrtf(var + 1e-5f) * g[t] + be[t];
}

// ---------------------------------------------------------------------------
// softmax over L*P = 16 logits, one thread per (b,q,h); in-place
// ---------------------------------------------------------------------------
__global__ void softmax16(float* __restrict__ aw) {
  long i = (long)blockIdx.x * blockDim.x + threadIdx.x;
  if (i >= (long)ROWS * HH) return;
  float* p = aw + i * 16;
  float mx = -1e30f;
#pragma unroll
  for (int j = 0; j < 16; ++j) mx = fmaxf(mx, p[j]);
  float s = 0.0f;
  float e[16];
#pragma unroll
  for (int j = 0; j < 16; ++j) { e[j] = __expf(p[j] - mx); s += e[j]; }
  float inv = 1.0f / s;
#pragma unroll
  for (int j = 0; j < 16; ++j) p[j] = e[j] * inv;
}

// ---------------------------------------------------------------------------
// Deformable bilinear sampling. One wave per (b,q,h); lane == Dh channel, so
// the 4-corner gathers are 32-lane coalesced over the channel dimension.
// Border handling: clamped index + weight zeroing (no exec-mask divergence).
// ---------------------------------------------------------------------------
__global__ void deform_sample(const float* __restrict__ qpos,
                              const float* __restrict__ off,
                              const float* __restrict__ aw,
                              const float* __restrict__ v0,
                              const float* __restrict__ v1,
                              const float* __restrict__ v2,
                              const float* __restrict__ v3,
                              float* __restrict__ ca) {
  int wid = blockIdx.x * (blockDim.x >> 5) + (threadIdx.x >> 5);
  int lane = threadIdx.x & 31;
  if (wid >= ROWS * HH) return;
  int h = wid % HH;
  long bq = wid / HH;
  int b = (int)(bq / QQ);

  const float px = qpos[bq * 2 + 0];
  const float py = qpos[bq * 2 + 1];
  const float* vbase[4] = {v0, v1, v2, v3};
  const int hs[4] = {92, 46, 23, 12};
  const int ws[4] = {92, 46, 23, 12};

  float acc = 0.0f;
#pragma unroll
  for (int l = 0; l < LL; ++l) {
    const int Hl = hs[l], Wl = ws[l];
    const float* vb = vbase[l] + (long)b * Hl * Wl * DD + h * DH + lane;
#pragma unroll
    for (int p = 0; p < PP; ++p) {
      long ob = bq * 256 + h * 32 + l * 8 + p * 2;
      float lx = px + off[ob]     / (float)Wl;
      float ly = py + off[ob + 1] / (float)Hl;
      float w = aw[bq * 128 + h * 16 + l * 4 + p];
      float fx = lx * Wl - 0.5f;
      float fy = ly * Hl - 0.5f;
      float x0 = floorf(fx), y0 = floorf(fy);
      float wx = fx - x0, wy = fy - y0;
      int xi = (int)x0, yi = (int)y0;
      float s = 0.0f;
#pragma unroll
      for (int dy = 0; dy < 2; ++dy)
#pragma unroll
        for (int dx = 0; dx < 2; ++dx) {
          int xx = xi + dx, yy = yi + dy;
          float okf = ((xx >= 0) & (xx < Wl) & (yy >= 0) & (yy < Hl)) ? 1.0f : 0.0f;
          int xc = xx < 0 ? 0 : (xx > Wl - 1 ? Wl - 1 : xx);
          int yc = yy < 0 ? 0 : (yy > Hl - 1 ? Hl - 1 : yy);
          float vv = vb[((long)yc * Wl + xc) * DD];
          float ww = (dx ? wx : 1.0f - wx) * (dy ? wy : 1.0f - wy) * okf;
          s += vv * ww;
        }
      acc += w * s;
    }
  }
  ca[bq * 256 + h * DH + lane] = acc;
}

// ---------------------------------------------------------------------------
// Host orchestration
// ---------------------------------------------------------------------------
extern "C" void kernel_launch(void* const* d_in, const int* in_sizes, int n_in,
                              void* d_out, int out_size, void* d_ws, size_t ws_size,
                              hipStream_t stream) {
  const float* query  = (const float*)d_in[0];
  const float* qpos   = (const float*)d_in[1];
  const float* mem[4] = {(const float*)d_in[2], (const float*)d_in[3],
                         (const float*)d_in[4], (const float*)d_in[5]};
  const float* in_w   = (const float*)d_in[6];
  const float* in_b   = (const float*)d_in[7];
  const float* out_w  = (const float*)d_in[8];
  const float* out_b  = (const float*)d_in[9];
  const float* ln1g   = (const float*)d_in[10];
  const float* ln1b   = (const float*)d_in[11];
  const float* val_w  = (const float*)d_in[12];
  const float* val_b  = (const float*)d_in[13];
  const float* off_w  = (const float*)d_in[14];
  const float* off_b  = (const float*)d_in[15];
  const float* aw_w   = (const float*)d_in[16];
  const float* aw_b   = (const float*)d_in[17];
  const float* co_w   = (const float*)d_in[18];
  const float* co_b   = (const float*)d_in[19];
  const float* ln2g   = (const float*)d_in[20];
  const float* ln2b   = (const float*)d_in[21];
  const float* ff1w   = (const float*)d_in[22];
  const float* ff1b   = (const float*)d_in[23];
  const float* ff2w   = (const float*)d_in[24];
  const float* ff2b   = (const float*)d_in[25];
  const float* ln3g   = (const float*)d_in[26];
  const float* ln3b   = (const float*)d_in[27];

  float* ws = (float*)d_ws;
  const long S = (long)ROWS * DD;   // 2,048,000
  float* PE   = ws;                 // [ROWS,256]
  float* QKB  = ws + S;             // query + pe
  float* QK2  = ws + 2 * S;         // fused q|k proj   [ROWS,512]
  float* VB   = ws + 4 * S;         // v proj           [ROWS,256]
  float* SA   = ws + 5 * S;         // attention output
  float* SAP  = ws + 6 * S;         // out-proj
  float* X1   = ws + 7 * S;
  float* QC   = ws + 8 * S;
  float* OFF  = ws + 9 * S;         // [ROWS,256]
  float* AWB  = ws + 10 * S;        // [ROWS,128]
  float* CA   = ws + 10 * S + S / 2;
  float* CAP  = ws + 11 * S + S / 2;
  float* X2   = ws + 12 * S + S / 2;
  float* FF1  = ws + 13 * S + S / 2;           // [ROWS,2048] = 8S
  float* FF2  = ws + 21 * S + S / 2;
  float* VAL  = ws + 22 * S + S / 2;           // per-level projected values
  float* VL[4];
  {
    long cur = 0;
    for (int l = 0; l < 4; ++l) {
      VL[l] = VAL + cur;
      cur += (long)BB * SH[l] * SW[l] * DD;
    }
  }

  dim3 gblk(128);   // 4 waves = 4 m-tiles, each wave does a 16x64 strip
  dim3 wave(32);

  // 1) positional embedding + qk
  pos_embed<<<dim3((S + 255) / 256), dim3(256), 0, stream>>>(query, qpos, PE, QKB);
  // 2) fused q|k projection (N=512) and v projection
  wmma_gemm<<<dim3(512 / 64, ROWS / 64, 1), gblk, 0, stream>>>(
      QKB, DD, 1, 0, in_w, in_b, QK2, 512, 0, ROWS, 512, DD, 0);
  wmma_gemm<<<dim3(256 / 64, ROWS / 64, 1), gblk, 0, stream>>>(
      query, DD, 1, 0, in_w + (long)512 * DD, in_b + 512, VB, 256, 0, ROWS, 256, DD, 0);
  // 3) flash attention (one wave per (b,h,qtile))
  flash_attn<<<dim3((QQ + 15) / 16, BB * HH), wave, 0, stream>>>(QK2, VB, SA);
  // 4) out-proj + LN1
  wmma_gemm<<<dim3(256 / 64, ROWS / 64, 1), gblk, 0, stream>>>(
      SA, DD, 1, 0, out_w, out_b, SAP, 256, 0, ROWS, 256, DD, 0);
  ln_res<<<dim3(ROWS), dim3(256), 0, stream>>>(query, SAP, ln1g, ln1b, X1);
  // 5) qc = x + pe; offset & weight heads
  add_vec<<<dim3((S + 255) / 256), dim3(256), 0, stream>>>(X1, PE, QC, S);
  wmma_gemm<<<dim3(256 / 64, ROWS / 64, 1), gblk, 0, stream>>>(
      QC, DD, 1, 0, off_w, off_b, OFF, 256, 0, ROWS, 256, DD, 0);
  wmma_gemm<<<dim3(128 / 64, ROWS / 64, 1), gblk, 0, stream>>>(
      QC, DD, 1, 0, aw_w, aw_b, AWB, 128, 0, ROWS, 128, DD, 0);
  softmax16<<<dim3((ROWS * HH + 255) / 256), dim3(256), 0, stream>>>(AWB);
  // 6) per-level value projection, reading mem[B,D,H,W] via strides (A[hw,d])
  for (int l = 0; l < 4; ++l) {
    int HW = SH[l] * SW[l];
    wmma_gemm<<<dim3(256 / 64, (HW + 63) / 64, BB), gblk, 0, stream>>>(
        mem[l], 1, HW, (long)DD * HW, val_w, val_b,
        VL[l], 256, (long)HW * 256, HW, 256, DD, 0);
  }
  // 7) deformable sampling (8 waves / block)
  deform_sample<<<dim3((ROWS * HH + 7) / 8), dim3(256), 0, stream>>>(
      qpos, OFF, AWB, VL[0], VL[1], VL[2], VL[3], CA);
  // 8) co-proj + LN2
  wmma_gemm<<<dim3(256 / 64, ROWS / 64, 1), gblk, 0, stream>>>(
      CA, DD, 1, 0, co_w, co_b, CAP, 256, 0, ROWS, 256, DD, 0);
  ln_res<<<dim3(ROWS), dim3(256), 0, stream>>>(X1, CAP, ln2g, ln2b, X2);
  // 9) FFN + LN3 -> d_out
  wmma_gemm<<<dim3(FF / 64, ROWS / 64, 1), gblk, 0, stream>>>(
      X2, DD, 1, 0, ff1w, ff1b, FF1, FF, 0, ROWS, FF, DD, 1);
  wmma_gemm<<<dim3(256 / 64, ROWS / 64, 1), gblk, 0, stream>>>(
      FF1, FF, 1, 0, ff2w, ff2b, FF2, 256, 0, ROWS, 256, FF, 0);
  ln_res<<<dim3(ROWS), dim3(256), 0, stream>>>(X2, FF2, ln3g, ln3b, (float*)d_out);
}
